// RNNBlock_57329223467124
// MI455X (gfx1250) — matrix operand order repaired and checked by reading, not compiled, not run on hardware
//
#include <hip/hip_runtime.h>
#include <hip/hip_bf16.h>

// Problem constants (from reference): B=16, L=2048, D=512, H=512
#define BB 16
#define LSEQ 2048
#define DD 512
#define HH 512
#define NSPLIT 8   // workgroups per scan direction

typedef __bf16 bf16;
typedef __attribute__((ext_vector_type(16))) __bf16 bf16x16;
typedef __attribute__((ext_vector_type(8)))  float  f32x8;
typedef __attribute__((ext_vector_type(4)))  unsigned int u32x4;

union FragU { bf16x16 v; u32x4 q[2]; };

// A fragment (16x32 bf16, MxK). Caller passes p = &A[m_row][k0 + (lane>>4)*16].
// Lane holds contiguous K run [half*16, half*16+16).
__device__ __forceinline__ bf16x16 load_frag_a(const bf16* p) {
  FragU f;
  f.q[0] = *reinterpret_cast<const u32x4*>(p);
  f.q[1] = *reinterpret_cast<const u32x4*>(p + 8);
  return f.v;
}

// B fragment (32x16 bf16, KxN) from transposed weights Bt[n][k].
// Caller passes p = &Bt[n_col][k0 + (lane>>4)*16]; 16 contiguous bf16.
__device__ __forceinline__ bf16x16 load_frag_b(const bf16* p) {
  FragU f;
  f.q[0] = *reinterpret_cast<const u32x4*>(p);
  f.q[1] = *reinterpret_cast<const u32x4*>(p + 8);
  return f.v;
}

__device__ __forceinline__ f32x8 wmma_bf16(bf16x16 a, bf16x16 b, f32x8 c) {
  return __builtin_amdgcn_wmma_f32_16x16x32_bf16(false, a, false, b, (short)0, c,
                                                 false, false);
}

// ---------------- elementwise fp32 -> bf16 ----------------
__global__ void cvt_f32_bf16_kernel(const float* __restrict__ src,
                                    bf16* __restrict__ dst, size_t n) {
  size_t i = (size_t)blockIdx.x * blockDim.x + threadIdx.x;
  if (i < n) dst[i] = (bf16)src[i];
}

// ---------------- transpose + convert: dst[c*R + r] = src[r*C + c] ----------------
__global__ void transpose_cvt_kernel(const float* __restrict__ src,
                                     bf16* __restrict__ dst, int R, int C) {
  int i = blockIdx.x * blockDim.x + threadIdx.x;
  if (i >= R * C) return;
  int r = i / C, c = i % C;
  dst[(size_t)c * R + r] = (bf16)src[i];
}

// ---------------- zero the scan-sync state (every launch -> deterministic) -------
__global__ void init_sync_kernel(bf16* __restrict__ hg, unsigned* __restrict__ flags) {
  int i = blockIdx.x * blockDim.x + threadIdx.x;
  if (i < 2 * 2 * BB * HH) hg[i] = (bf16)0.f;   // both h double-buffers, both dirs
  if (i < 2 * LSEQ) flags[i] = 0u;              // per-step arrival counters
}

// ---------------- generic WMMA GEMM ----------------
// C[crow(m)][c_coloff + n] = sum_k A[arow(m)][k] * Bt[n][k] + bias[n]
// amap/cmap: 0 => identity row; 1 => row = (m&15)*LSEQ + (m>>4)  (l*16+b <-> b*L+l swizzle)
// WG: 256 threads = 8 waves, 4 (M) x 2 (N); wave tile 32x64; block tile 128x128.
__global__ __launch_bounds__(256) void gemm_bf16_kernel(
    const bf16* __restrict__ A, const bf16* __restrict__ Bt,
    const float* __restrict__ bias, void* __restrict__ Cout,
    int K, int amap, int cmap, int c_is_f32, int ldC, int c_coloff) {
  const int tid  = threadIdx.x;
  const int lane = tid & 31;
  const int wave = tid >> 5;
  const int wm = wave & 3, wn = wave >> 2;
  const int half = lane >> 4, l15 = lane & 15;

  const int m_base = blockIdx.x * 128 + wm * 32;
  const int n_base = blockIdx.y * 128 + wn * 64;

  f32x8 acc[2][4];
#pragma unroll
  for (int i = 0; i < 2; ++i)
#pragma unroll
    for (int j = 0; j < 4; ++j) {
      f32x8 z = {0.f, 0.f, 0.f, 0.f, 0.f, 0.f, 0.f, 0.f};
      acc[i][j] = z;
    }

  const bf16* aptr[2];
#pragma unroll
  for (int i = 0; i < 2; ++i) {
    int m = m_base + i * 16 + l15;
    size_t arow = (amap == 0) ? (size_t)m
                              : ((size_t)(m & 15) * LSEQ + (size_t)(m >> 4));
    aptr[i] = A + arow * (size_t)K + half * 16;
  }
  const bf16* bptr[4];
#pragma unroll
  for (int j = 0; j < 4; ++j)
    bptr[j] = Bt + (size_t)(n_base + j * 16 + l15) * (size_t)K + half * 16;

  for (int k0 = 0; k0 < K; k0 += 32) {
    bf16x16 a0 = load_frag_a(aptr[0] + k0);
    bf16x16 a1 = load_frag_a(aptr[1] + k0);
    bf16x16 b0 = load_frag_b(bptr[0] + k0);
    bf16x16 b1 = load_frag_b(bptr[1] + k0);
    bf16x16 b2 = load_frag_b(bptr[2] + k0);
    bf16x16 b3 = load_frag_b(bptr[3] + k0);
    acc[0][0] = wmma_bf16(a0, b0, acc[0][0]);
    acc[0][1] = wmma_bf16(a0, b1, acc[0][1]);
    acc[0][2] = wmma_bf16(a0, b2, acc[0][2]);
    acc[0][3] = wmma_bf16(a0, b3, acc[0][3]);
    acc[1][0] = wmma_bf16(a1, b0, acc[1][0]);
    acc[1][1] = wmma_bf16(a1, b1, acc[1][1]);
    acc[1][2] = wmma_bf16(a1, b2, acc[1][2]);
    acc[1][3] = wmma_bf16(a1, b3, acc[1][3]);
  }

#pragma unroll
  for (int j = 0; j < 4; ++j) {
    int col = n_base + j * 16 + l15;
    float bv = bias ? bias[col] : 0.f;
#pragma unroll
    for (int i = 0; i < 2; ++i) {
#pragma unroll
      for (int g = 0; g < 8; ++g) {
        int m = m_base + i * 16 + g + half * 8;
        size_t crow = (cmap == 0) ? (size_t)m
                                  : ((size_t)(m & 15) * LSEQ + (size_t)(m >> 4));
        float v = acc[i][j][g] + bv;
        if (c_is_f32)
          ((float*)Cout)[crow * (size_t)ldC + c_coloff + col] = v;
        else
          ((bf16*)Cout)[crow * (size_t)ldC + c_coloff + col] = (bf16)v;
      }
    }
  }
}

// ---------------- distributed recurrent scan: h = tanh(pre[l] + h @ Wh) ----------
// grid = (NSPLIT, 2 dirs); block = 128 threads = 4 waves (1 wave per SIMD).
// Each wave owns ONE 16-column tile and keeps its 16 B-fragments of WhT
// (128 VGPRs) resident for all 2048 steps -> zero weight traffic in the loop.
// h is double-buffered in global memory (L2-resident, 16 KB each); step barrier
// is a per-step arrival counter (race-free, no reset needed within the kernel).
__global__ __launch_bounds__(128) void rnn_scan_dist_kernel(
    const bf16* __restrict__ pre_f, const bf16* __restrict__ pre_b,
    const bf16* __restrict__ WhT_f, const bf16* __restrict__ WhT_b,
    bf16* __restrict__ hs_f, bf16* __restrict__ hs_b,
    bf16* __restrict__ hg, unsigned* __restrict__ flags) {
  const int g   = blockIdx.x;   // 0..NSPLIT-1
  const int dir = blockIdx.y;   // 0 fwd, 1 bwd
  const bf16* pre = dir ? pre_b : pre_f;
  const bf16* WhT = dir ? WhT_b : WhT_f;
  bf16* hs = dir ? hs_b : hs_f;
  bf16* hbuf = hg + (size_t)dir * 2 * (BB * HH);
  unsigned* flag = flags + (size_t)dir * LSEQ;

  const int tid  = threadIdx.x;
  const int lane = tid & 31;
  const int wave = tid >> 5;                 // 0..3
  const int half = lane >> 4, l15 = lane & 15;
  const int nt  = g * 4 + wave;              // global n-tile 0..31
  const int col = nt * 16 + l15;

  // Pin this wave's slice of WhT in registers: 16 B-fragments (K=512).
  bf16x16 bw[16];
#pragma unroll
  for (int kt = 0; kt < 16; ++kt)
    bw[kt] = load_frag_b(WhT + (size_t)(nt * 16 + l15) * HH + half * 16 + kt * 32);

  for (int t = 0; t < LSEQ; ++t) {
    const int l = dir ? (LSEQ - 1 - t) : t;
    const bf16* hread  = hbuf + ((t + 1) & 1) * (BB * HH);  // h(t-1)
    bf16*       hwrite = hbuf + (t & 1) * (BB * HH);        // h(t)

    // Issue pre[l] loads BEFORE the barrier spin so their latency hides under it.
    const size_t prebase = (size_t)l * (BB * HH);
    float pv[8];
#pragma unroll
    for (int gg = 0; gg < 8; ++gg)
      pv[gg] = (float)pre[prebase + (size_t)(gg + half * 8) * HH + col];
    if (t + 1 < LSEQ) {  // prefetch next step's pre slice for this WG
      const int ln = dir ? (LSEQ - 2 - t) : (t + 1);
      if (tid < BB)
        __builtin_prefetch(pre + (size_t)ln * (BB * HH) + (size_t)tid * HH + g * 64,
                           0, 1);
    }

    if (t > 0) {
      if (tid == 0) {
        while (__hip_atomic_load(&flag[t - 1], __ATOMIC_ACQUIRE,
                                 __HIP_MEMORY_SCOPE_AGENT) < (unsigned)NSPLIT)
          __builtin_amdgcn_s_sleep(0);
      }
      __syncthreads();
      __threadfence();  // acquire side: make all WGs' h(t-1) stores visible
    }

    // acc = h(t-1) @ Wh[:, tile]; FOUR independent chains -> dependent depth 4.
    const bf16* ar = hread + l15 * HH + half * 16;
    f32x8 ac0 = {0.f,0.f,0.f,0.f,0.f,0.f,0.f,0.f};
    f32x8 ac1 = {0.f,0.f,0.f,0.f,0.f,0.f,0.f,0.f};
    f32x8 ac2 = {0.f,0.f,0.f,0.f,0.f,0.f,0.f,0.f};
    f32x8 ac3 = {0.f,0.f,0.f,0.f,0.f,0.f,0.f,0.f};
#pragma unroll
    for (int kt = 0; kt < 16; kt += 4) {
      bf16x16 a0 = load_frag_a(ar + kt * 32);
      bf16x16 a1 = load_frag_a(ar + kt * 32 + 32);
      bf16x16 a2 = load_frag_a(ar + kt * 32 + 64);
      bf16x16 a3 = load_frag_a(ar + kt * 32 + 96);
      ac0 = wmma_bf16(a0, bw[kt],     ac0);
      ac1 = wmma_bf16(a1, bw[kt + 1], ac1);
      ac2 = wmma_bf16(a2, bw[kt + 2], ac2);
      ac3 = wmma_bf16(a3, bw[kt + 3], ac3);
    }

    // Activation + publish h(t) slice and hs.
#pragma unroll
    for (int gg = 0; gg < 8; ++gg) {
      const int r = gg + half * 8;
      bf16 hv = (bf16)tanhf((ac0[gg] + ac1[gg]) + (ac2[gg] + ac3[gg]) + pv[gg]);
      hwrite[(size_t)r * HH + col] = hv;
      hs[prebase + (size_t)r * HH + col] = hv;
    }
    __threadfence();   // release side: flush this thread's h stores toward L2
    __syncthreads();   // whole WG done writing its slice
    if (tid == 0)
      __hip_atomic_fetch_add(&flag[t], 1u, __ATOMIC_RELEASE,
                             __HIP_MEMORY_SCOPE_AGENT);
  }
}

extern "C" void kernel_launch(void* const* d_in, const int* in_sizes, int n_in,
                              void* d_out, int out_size, void* d_ws, size_t ws_size,
                              hipStream_t stream) {
  (void)in_sizes; (void)n_in; (void)out_size; (void)ws_size;
  const float* x    = (const float*)d_in[0];
  // d_in[1] = c (unused by the reference)
  const float* Wx_f = (const float*)d_in[2];
  const float* Wh_f = (const float*)d_in[3];
  const float* b_f  = (const float*)d_in[4];
  const float* Wo_f = (const float*)d_in[5];
  const float* bo_f = (const float*)d_in[6];
  const float* Wx_b = (const float*)d_in[7];
  const float* Wh_b = (const float*)d_in[8];
  const float* b_b  = (const float*)d_in[9];
  const float* Wo_b = (const float*)d_in[10];
  const float* bo_b = (const float*)d_in[11];
  const float* Wd   = (const float*)d_in[12];
  const float* bd   = (const float*)d_in[13];

  char* w = (char*)d_ws;
  size_t off = 0;
  const size_t SZ_X   = (size_t)BB * LSEQ * DD * 2;      // 32 MB bf16
  const size_t SZ_W   = (size_t)DD * HH * 2;             // 512 KB bf16
  const size_t SZ_WD  = (size_t)(2 * DD) * DD * 2;       // 1 MB bf16
  const size_t SZ_LBH = (size_t)LSEQ * BB * HH * 2;      // 32 MB bf16
  const size_t SZ_CAT = (size_t)BB * LSEQ * 2 * DD * 2;  // 64 MB bf16
  const size_t SZ_HG  = (size_t)2 * 2 * BB * HH * 2;     // 64 KB
  const size_t SZ_FLG = (size_t)2 * LSEQ * 4;            // 16 KB

  bf16* xb    = (bf16*)(w + off); off += SZ_X;
  bf16* WxTf  = (bf16*)(w + off); off += SZ_W;
  bf16* WhTf  = (bf16*)(w + off); off += SZ_W;
  bf16* WoTf  = (bf16*)(w + off); off += SZ_W;
  bf16* WxTb  = (bf16*)(w + off); off += SZ_W;
  bf16* WhTb  = (bf16*)(w + off); off += SZ_W;
  bf16* WoTb  = (bf16*)(w + off); off += SZ_W;
  bf16* WdT   = (bf16*)(w + off); off += SZ_WD;
  bf16* pre_f = (bf16*)(w + off); off += SZ_LBH;
  bf16* pre_b = (bf16*)(w + off); off += SZ_LBH;
  bf16* hs_f  = (bf16*)(w + off); off += SZ_LBH;
  bf16* hs_b  = (bf16*)(w + off); off += SZ_LBH;
  bf16* ycat  = (bf16*)(w + off); off += SZ_CAT;
  bf16* hg    = (bf16*)(w + off); off += SZ_HG;
  unsigned* flags = (unsigned*)(w + off); off += SZ_FLG;

  // 0) zero the scan-sync state (must happen every launch for graph replays)
  init_sync_kernel<<<(2 * 2 * BB * HH + 255) / 256, 256, 0, stream>>>(hg, flags);

  // 1) x -> bf16
  {
    size_t n = (size_t)BB * LSEQ * DD;
    cvt_f32_bf16_kernel<<<(unsigned)((n + 255) / 256), 256, 0, stream>>>(x, xb, n);
  }
  // 2) weights -> transposed bf16 (Bt[n][k] = W[k][n])
  {
    int n1 = DD * HH;
    unsigned g1 = (n1 + 255) / 256;
    transpose_cvt_kernel<<<g1, 256, 0, stream>>>(Wx_f, WxTf, DD, HH);
    transpose_cvt_kernel<<<g1, 256, 0, stream>>>(Wh_f, WhTf, HH, HH);
    transpose_cvt_kernel<<<g1, 256, 0, stream>>>(Wo_f, WoTf, HH, DD);
    transpose_cvt_kernel<<<g1, 256, 0, stream>>>(Wx_b, WxTb, DD, HH);
    transpose_cvt_kernel<<<g1, 256, 0, stream>>>(Wh_b, WhTb, HH, HH);
    transpose_cvt_kernel<<<g1, 256, 0, stream>>>(Wo_b, WoTb, HH, DD);
    int n2 = 2 * DD * DD;
    transpose_cvt_kernel<<<(unsigned)((n2 + 255) / 256), 256, 0, stream>>>(
        Wd, WdT, 2 * DD, DD);
  }
  const int M = BB * LSEQ;  // 32768
  dim3 blk(256);
  dim3 grid(M / 128, HH / 128);  // 256 x 4

  // 3) pre = x @ Wx + b   (M rows indexed m = l*16+b; A row = b*L+l -> amap=1)
  gemm_bf16_kernel<<<grid, blk, 0, stream>>>(xb, WxTf, b_f, pre_f,
                                             DD, 1, 0, 0, HH, 0);
  gemm_bf16_kernel<<<grid, blk, 0, stream>>>(xb, WxTb, b_b, pre_b,
                                             DD, 1, 0, 0, HH, 0);

  // 4) distributed sequential scans: 8 WGs per direction, register-pinned Wh
  {
    dim3 sgrid(NSPLIT, 2);
    rnn_scan_dist_kernel<<<sgrid, 128, 0, stream>>>(pre_f, pre_b, WhTf, WhTb,
                                                    hs_f, hs_b, hg, flags);
  }

  // 5) y_fwd/y_bwd = hs @ Wo + bo, written into concat buffer rows b*L+l (cmap=1)
  gemm_bf16_kernel<<<grid, blk, 0, stream>>>(hs_f, WoTf, bo_f, ycat,
                                             HH, 0, 1, 0, 2 * DD, 0);
  gemm_bf16_kernel<<<grid, blk, 0, stream>>>(hs_b, WoTb, bo_b, ycat,
                                             HH, 0, 1, 0, 2 * DD, DD);

  // 6) out = ycat @ Wd + bd  (fp32 output)
  gemm_bf16_kernel<<<grid, blk, 0, stream>>>(ycat, WdT, bd, d_out,
                                             2 * DD, 0, 0, 1, DD, 0);
}
